// CapsNet_38414187495723
// MI455X (gfx1250) — matrix operand order repaired and checked by reading, not compiled, not run on hardware
//
#include <hip/hip_runtime.h>

typedef __attribute__((ext_vector_type(16))) __bf16 v16bf;
typedef __attribute__((ext_vector_type(8)))  float  v8f;
typedef __attribute__((ext_vector_type(2)))  float  v2f;
typedef __attribute__((ext_vector_type(4)))  unsigned int u32x4;
typedef __attribute__((ext_vector_type(8)))  int i32x8;
typedef __attribute__((ext_vector_type(4)))  int i32x4;
typedef unsigned short ushort_t;

#define KTOT 20736
#define NPRIM 1152
#define BATCH 256

__device__ inline ushort_t f2bf(float f) {
  unsigned u = __builtin_bit_cast(unsigned, f);
  u = u + 0x7FFFu + ((u >> 16) & 1u);   // round-to-nearest-even
  return (ushort_t)(u >> 16);
}

union Frag32B { uint4 q[2]; v16bf v; };

__device__ inline v16bf load_frag(const ushort_t* p0, const ushort_t* p1) {
  Frag32B f;
  f.q[0] = *(const uint4*)p0;
  f.q[1] = *(const uint4*)p1;
  return f.v;
}

// ---------------------------------------------------------------------------
// Kernel 1: conv1 9x9 s1 (1 -> 256 ch) + ReLU, output bf16 [B,20,20,256]
// One block per output pixel; 256 threads = output channels; patch in LDS.
// ---------------------------------------------------------------------------
__global__ __launch_bounds__(256) void k_conv1(const float* __restrict__ x,
                                               const float* __restrict__ w1,
                                               const float* __restrict__ b1,
                                               ushort_t* __restrict__ act1) {
  int pix = blockIdx.x;               // 0 .. B*400-1
  int b  = pix / 400;
  int r  = pix % 400;
  int oy = r / 20, ox = r % 20;
  __shared__ float sx[81];
  int t = threadIdx.x;
  if (t < 81) {
    int ky = t / 9, kx = t % 9;
    sx[t] = x[(b * 28 + oy + ky) * 28 + ox + kx];
  }
  __syncthreads();
  float acc = b1[t];
  #pragma unroll 9
  for (int q = 0; q < 81; ++q) acc = fmaf(sx[q], w1[q * 256 + t], acc);
  acc = fmaxf(acc, 0.f);
  act1[(size_t)pix * 256 + t] = f2bf(acc);
}

// ---------------------------------------------------------------------------
// Kernel 2: transpose+convert pc_w f32 [K=20736][N=256] -> bf16 Bt [256][20736]
// ---------------------------------------------------------------------------
__global__ __launch_bounds__(256) void k_wt(const float* __restrict__ pcw,
                                            ushort_t* __restrict__ Bt) {
  __shared__ float tile[32][33];
  int kb = blockIdx.x * 32, nb = blockIdx.y * 32;
  int tx = threadIdx.x & 31, ty = threadIdx.x >> 5;   // 32 x 8
  for (int yy = ty; yy < 32; yy += 8)
    tile[yy][tx] = pcw[(size_t)(kb + yy) * 256 + nb + tx];
  __syncthreads();
  for (int yy = ty; yy < 32; yy += 8)
    Bt[(size_t)(nb + yy) * KTOT + kb + tx] = f2bf(tile[tx][yy]);
}

// ---------------------------------------------------------------------------
// Kernel 3: primary-caps conv as implicit GEMM, bf16 WMMA 16x16x32, f32 accum.
// C[9216][256] = im2col(act1) x Bt^T, + pc_b.  Wave tile 64x32 (8 accumulators),
// block = 8 waves -> 128x128 C block.  Output U == u reshaped [B,1152,8] flat.
// ---------------------------------------------------------------------------
__global__ __launch_bounds__(256) void k_pcgemm(const ushort_t* __restrict__ A,
                                                const ushort_t* __restrict__ Bt,
                                                const float* __restrict__ pcb,
                                                float* __restrict__ U) {
  int wave = threadIdx.x >> 5;
  int lane = threadIdx.x & 31;
  int wm = wave >> 2, wn = wave & 3;
  int m0 = blockIdx.x * 128 + wm * 64;
  int n0 = blockIdx.y * 128 + wn * 32;
  int lh = lane >> 4;                 // half-wave select
  int lm = lane & 15;

  int rowbase[4];
  #pragma unroll
  for (int s = 0; s < 4; ++s) {
    int m = m0 + s * 16 + lm;
    int b = m / 36, p = m % 36;
    int oy = p / 6, ox = p % 6;
    rowbase[s] = ((b * 20 + oy * 2) * 20 + ox * 2) * 256;
  }
  const ushort_t* brow0 = Bt + (size_t)(n0 + lm) * KTOT + lh * 16;
  const ushort_t* brow1 = brow0 + (size_t)16 * KTOT;

  v8f zero = {0.f, 0.f, 0.f, 0.f, 0.f, 0.f, 0.f, 0.f};
  v8f acc[4][2];
  #pragma unroll
  for (int s = 0; s < 4; ++s)
    #pragma unroll
    for (int t = 0; t < 2; ++t) acc[s][t] = zero;

  for (int kb = 0; kb < KTOT; kb += 32) {
    int kxy = kb >> 8;                 // (ky*9+kx), 256 % 32 == 0 so no straddle
    int ci0 = kb & 255;
    int ky = kxy / 9, kx = kxy % 9;
    int koff = (ky * 20 + kx) * 256 + ci0;

    // prefetch shared B panel two K-steps ahead (L2-resident, keeps L0 warm)
    if (kb + 64 < KTOT) {
      __builtin_prefetch(brow0 + kb + 64, 0, 1);
      __builtin_prefetch(brow1 + kb + 64, 0, 1);
    }

    v16bf af[4], bf[2];
    #pragma unroll
    for (int s = 0; s < 4; ++s) {
      const ushort_t* p = A + rowbase[s] + koff + lh * 16;
      af[s] = load_frag(p, p + 8);     // K contiguous 16 per half-wave
    }
    {
      const ushort_t* p0 = brow0 + kb;
      const ushort_t* p1 = brow1 + kb;
      bf[0] = load_frag(p0, p0 + 8);   // K contiguous 16 per half-wave
      bf[1] = load_frag(p1, p1 + 8);
    }
    #pragma unroll
    for (int s = 0; s < 4; ++s)
      #pragma unroll
      for (int t = 0; t < 2; ++t)
        acc[s][t] = __builtin_amdgcn_wmma_f32_16x16x32_bf16(
            false, af[s], false, bf[t], (short)0, acc[s][t], false, false);
  }

  #pragma unroll
  for (int s = 0; s < 4; ++s)
    #pragma unroll
    for (int t = 0; t < 2; ++t) {
      int n = n0 + t * 16 + lm;
      float bias = pcb[n];
      #pragma unroll
      for (int r = 0; r < 8; ++r) {
        int m = m0 + s * 16 + r + lh * 8;
        U[(size_t)m * 256 + n] = acc[s][t][r] + bias;
      }
    }
}

// ---------------------------------------------------------------------------
// Kernel 4: fused dynamic routing (3 iters) per batch element.
// u[b] (36KB) brought into LDS by the Tensor Data Mover (one D# per block,
// issued by wave 0, overlapped with logits init, s_wait_tensorcnt to join).
// u_hat recomputed from L2-resident W.  Writes v -> d_out, masked v_ -> ws.
// ---------------------------------------------------------------------------
__global__ __launch_bounds__(256) void k_route(const float* __restrict__ U,
                                               const float* __restrict__ W,
                                               const int* __restrict__ y,
                                               float* __restrict__ vout,
                                               float* __restrict__ vm) {
  extern __shared__ float sm[];
  float* su     = sm;                 // 9216
  float* blog   = su + 9216;          // 11520
  float* nmax   = blog + 11520;       // 1152
  float* nrcp   = nmax + 1152;        // 1152
  float* ss     = nrcp + 1152;        // 160
  float* sscale = ss + 160;           // 16

  int b = blockIdx.x, t = threadIdx.x;

  // --- TDM: LDS[su][0:9216) = U[b*9216 : (b+1)*9216)  (1 row x 9216 f32) ---
  if (t < 32) {                        // wave 0 only (uniform branch)
    unsigned long long ga = (unsigned long long)(uintptr_t)(U + (size_t)b * 9216);
    unsigned lds_addr = (unsigned)(unsigned long long)(uintptr_t)su;
    u32x4 g0;
    g0[0] = 1u;                                            // count=1, user mode
    g0[1] = lds_addr;                                      // lds_addr
    g0[2] = (unsigned)(ga & 0xFFFFFFFFu);                  // global_addr lo
    g0[3] = (unsigned)((ga >> 32) & 0x1FFFFFFu) | (2u << 30); // addr hi | type=2
    i32x8 g1;
    g1[0] = 0x20000;                   // workgroup_mask=0, data_size=2 (4B)
    g1[1] = (int)(9216u << 16);        // tensor_dim0[15:0] in [63:48]
    g1[2] = (int)(1u << 16);           // tensor_dim0 hi = 0, tensor_dim1 = 1
    g1[3] = (int)(9216u << 16);        // tensor_dim1 hi = 0, tile_dim0 = 9216
    g1[4] = 1;                         // tile_dim1 = 1, tile_dim2 = 0
    g1[5] = 9216;                      // tensor_dim0_stride lo
    g1[6] = 0;                         // stride hi, dim1_stride lo
    g1[7] = 0;                         // dim1_stride hi
    i32x4 gz = {0, 0, 0, 0};
#if __clang_major__ >= 23
    i32x8 gz8 = {0, 0, 0, 0, 0, 0, 0, 0};
    __builtin_amdgcn_tensor_load_to_lds(g0, g1, gz, gz, gz8, 0);
#else
    __builtin_amdgcn_tensor_load_to_lds(g0, g1, gz, gz, 0);
#endif
  }
  for (int i = t; i < 11520; i += 256) blog[i] = 0.f;   // overlaps with TDM
  if (t < 32) __builtin_amdgcn_s_wait_tensorcnt(0);
  __syncthreads();

  for (int it = 0; it < 3; ++it) {
    // per-primary-capsule softmax stats over the 10 routes
    for (int n = t; n < NPRIM; n += 256) {
      float m = blog[n * 10];
      #pragma unroll
      for (int k = 1; k < 10; ++k) m = fmaxf(m, blog[n * 10 + k]);
      float s = 0.f;
      #pragma unroll
      for (int k = 0; k < 10; ++k) s += __expf(blog[n * 10 + k] - m);
      nmax[n] = m;
      nrcp[n] = 1.f / s;
    }
    __syncthreads();
    // s[k,i] = sum_n c[n,k] * (W[n,k,i,:] . u[n,:])
    if (t < 160) {
      int k = t >> 4, i = t & 15;
      float acc = 0.f;
      for (int n = 0; n < NPRIM; ++n) {
        const float* wp = W + ((size_t)(n * 10 + k) * 16 + i) * 8;
        const float* up = su + n * 8;
        float uh = 0.f;
        #pragma unroll
        for (int j = 0; j < 8; ++j) uh = fmaf(wp[j], up[j], uh);
        float c = __expf(blog[n * 10 + k] - nmax[n]) * nrcp[n];
        acc = fmaf(c, uh, acc);
      }
      ss[t] = acc;
    }
    __syncthreads();
    if (t < 10) {                      // squash scale per secondary capsule
      float n2 = 0.f;
      #pragma unroll
      for (int i = 0; i < 16; ++i) { float v = ss[t * 16 + i]; n2 = fmaf(v, v, n2); }
      sscale[t] = n2 / (1.f + n2) / (sqrtf(n2) + 1e-7f);
    }
    __syncthreads();
    if (t < 160) ss[t] *= sscale[t >> 4];
    __syncthreads();
    if (it < 2) {                      // b += <u_hat, v>  (skip on last iter)
      for (int n = t; n < NPRIM; n += 256) {
        const float* up = su + n * 8;
        #pragma unroll 2
        for (int k = 0; k < 10; ++k) {
          const float* wp = W + (size_t)(n * 10 + k) * 128;
          float agg = 0.f;
          for (int i = 0; i < 16; ++i) {
            float uh = 0.f;
            #pragma unroll
            for (int j = 0; j < 8; ++j) uh = fmaf(wp[i * 8 + j], up[j], uh);
            agg = fmaf(uh, ss[k * 16 + i], agg);
          }
          blog[n * 10 + k] += agg;
        }
      }
      __syncthreads();
    }
  }
  if (t < 160) {
    vout[(size_t)b * 160 + t] = ss[t];
    int lbl = y[b];
    vm[(size_t)b * 160 + t] = ((t >> 4) == lbl) ? ss[t] : 0.f;
  }
}

// ---------------------------------------------------------------------------
// Kernel 5: decoder dense layer via exact-fp32 WMMA 16x16x4.
// out[256][N] = act(A[256][K] x Wt[K][N] + bias).  One wave per 16x16 tile.
// ---------------------------------------------------------------------------
template <int ACT>
__global__ __launch_bounds__(256) void k_dense(const float* __restrict__ A,
                                               const float* __restrict__ Wt,
                                               const float* __restrict__ bias,
                                               float* __restrict__ out,
                                               int K, int N) {
  int id = blockIdx.x * 8 + (threadIdx.x >> 5);
  int lane = threadIdx.x & 31;
  int ntiles = N >> 4;
  if (id >= 16 * ntiles) return;
  int tm = id & 15, tn = id >> 4;
  int lm = lane & 15, lh = lane >> 4;

  const float* arow = A + (size_t)(tm * 16 + lm) * K + lh * 2;
  const float* bcol = Wt + tn * 16 + lm;
  v8f acc = {0.f, 0.f, 0.f, 0.f, 0.f, 0.f, 0.f, 0.f};

  for (int k = 0; k < K; k += 4) {
    v2f a, bb;
    a.x = arow[k];                      // A[m][k + lh*2 .. +1]
    a.y = arow[k + 1];
    const float* bp = bcol + (size_t)(k + lh * 2) * N;
    bb.x = bp[0];                       // B[k + lh*2][n], B[k+1+lh*2][n]
    bb.y = bp[N];
    acc = __builtin_amdgcn_wmma_f32_16x16x4_f32(
        false, a, false, bb, (short)0, acc, false, false);
  }
  int n = tn * 16 + lm;
  float bs = bias[n];
  #pragma unroll
  for (int r = 0; r < 8; ++r) {
    int m = tm * 16 + r + lh * 8;
    float v = acc[r] + bs;
    v = (ACT == 0) ? fmaxf(v, 0.f) : 1.f / (1.f + __expf(-v));
    out[(size_t)m * N + n] = v;
  }
}

// ---------------------------------------------------------------------------
extern "C" void kernel_launch(void* const* d_in, const int* in_sizes, int n_in,
                              void* d_out, int out_size, void* d_ws, size_t ws_size,
                              hipStream_t stream) {
  (void)in_sizes; (void)n_in; (void)out_size; (void)ws_size;
  const float* x    = (const float*)d_in[0];
  const int*   y    = (const int*)  d_in[1];
  const float* c1w  = (const float*)d_in[2];
  const float* c1b  = (const float*)d_in[3];
  const float* pcw  = (const float*)d_in[4];
  const float* pcb  = (const float*)d_in[5];
  const float* w    = (const float*)d_in[6];
  const float* d1w  = (const float*)d_in[7];
  const float* d1b  = (const float*)d_in[8];
  const float* d2w  = (const float*)d_in[9];
  const float* d2b  = (const float*)d_in[10];
  const float* d3w  = (const float*)d_in[11];
  const float* d3b  = (const float*)d_in[12];

  char* ws = (char*)d_ws;
  ushort_t* act1 = (ushort_t*)ws;                       // 52,428,800 B
  ushort_t* Bt   = (ushort_t*)(ws + 52428800);          // 10,616,832 B
  float*    U    = (float*)(ws + 52428800 + 10616832);  // 9,437,184 B
  float*    vmB  = U + 2359296;                         // 163,840 B
  float*    h1   = vmB + 40960;                         // 524,288 B
  float*    h2   = h1 + 131072;                         // 1,048,576 B

  float* vout  = (float*)d_out;          // v: [256,1,10,16]
  float* recon = vout + 40960;           // recon: [256,784]

  k_conv1 <<<BATCH * 400, 256, 0, stream>>>(x, c1w, c1b, act1);
  k_wt    <<<dim3(KTOT / 32, 256 / 32), 256, 0, stream>>>(pcw, Bt);
  k_pcgemm<<<dim3(9216 / 128, 256 / 128), 256, 0, stream>>>(act1, Bt, pcb, U);
  k_route <<<BATCH, 256, 92864, stream>>>(U, w, y, vout, vmB);
  k_dense<0><<<64, 256, 0, stream>>>(vmB, d1w, d1b, h1, 160, 512);
  k_dense<0><<<128, 256, 0, stream>>>(h1, d2w, d2b, h2, 512, 1024);
  k_dense<1><<<98, 256, 0, stream>>>(h2, d3w, d3b, recon, 1024, 784);
}
